// PairwiseAttention_24824910971210
// MI455X (gfx1250) — compile-verified, matrix operand order checked by reading, not run on hardware
//
#include <hip/hip_runtime.h>

typedef __attribute__((ext_vector_type(16))) _Float16 v16h;
typedef __attribute__((ext_vector_type(8)))  _Float16 v8h;
typedef __attribute__((ext_vector_type(8)))  float    v8f;
typedef __attribute__((ext_vector_type(4)))  float    v4f;

#define B_  8
#define C_  256
#define CI_ 128
#define N_  4096

// Build a 16-bit A/B-matrix fragment (v16h -> 8 VGPRs) from a row-major source
// where the 32 K-values for this fragment's row live contiguously at rowbase+kbase.
// Per ISA 7.12.2 (16-bit A 16x32): lanes 0-15 hold K {0..7, 16..23},
// lanes 16-31 hold K {8..15, 24..31}; element i of v16h -> VGPR i/2 half i&1.
__device__ __forceinline__ v16h frag_from(const _Float16* rowbase, int kbase, int lane) {
  const int h = (lane >> 4) & 1;
  v8h a = *(const v8h*)(rowbase + kbase + 8 * h);
  v8h b = *(const v8h*)(rowbase + kbase + 16 + 8 * h);
  v16h r;
#pragma unroll
  for (int i = 0; i < 8; ++i) { r[i] = a[i]; r[i + 8] = b[i]; }
  return r;
}

// Async DMA of one 16-byte chunk from global memory into LDS (per lane).
// Tracked by ASYNCcnt; LDS address is wave-relative (low 32 bits of generic ptr).
__device__ __forceinline__ void async_copy_b128(unsigned lds_off, const void* gaddr) {
  asm volatile("global_load_async_to_lds_b128 %0, %1, off"
               :: "v"(lds_off), "v"(gaddr) : "memory");
}

__device__ __forceinline__ unsigned lds_off_of(const void* p) {
  return (unsigned)(size_t)p;  // generic LDS addr truncates to wave-relative offset
}

// ---------------------------------------------------------------------------
// Kernel 1: fused Q/K/V 1x1-conv projections as a f16 WMMA GEMM.
// Out tile per block: 128 pixels x 64 output channels. K-dim = C_ = 256.
// Q,K written (B,N,CI) pixel-major f16; V written (B,C,N) channel-major f16.
// ---------------------------------------------------------------------------
__global__ __launch_bounds__(256) void qkv_proj(
    const float* __restrict__ x,
    const float* __restrict__ Wq, const float* __restrict__ bq,
    const float* __restrict__ Wk, const float* __restrict__ bk,
    const float* __restrict__ Wv, const float* __restrict__ bv,
    _Float16* __restrict__ Qo, _Float16* __restrict__ Ko, _Float16* __restrict__ Vo) {
  __shared__ _Float16 Ald[128 * 40];  // 128 pixels x 32 ch, pitch 40 halves (16B rows)
  __shared__ _Float16 Bld[64 * 40];   // 64 outch x 32 ch

  const int tid  = threadIdx.x;
  const int lane = tid & 31;
  const int wave = tid >> 5;
  const int p0 = blockIdx.x * 128;
  const int o0 = blockIdx.y * 64;
  const int b  = blockIdx.z;

  const float* W; const float* bias; _Float16* dst; int crel; bool vmaj;
  if (o0 < 128)      { W = Wq; bias = bq; dst = Qo; crel = o0;       vmaj = false; }
  else if (o0 < 256) { W = Wk; bias = bk; dst = Ko; crel = o0 - 128; vmaj = false; }
  else               { W = Wv; bias = bv; dst = Vo; crel = o0 - 256; vmaj = true;  }

  const v8f zero = {0, 0, 0, 0, 0, 0, 0, 0};
  v8f acc[4];
#pragma unroll
  for (int t = 0; t < 4; ++t) acc[t] = zero;

  for (int c0 = 0; c0 < C_; c0 += 32) {
    // Stage A = x^T tile (transpose channel-major x through LDS, f32 -> f16).
#pragma unroll
    for (int it = 0; it < 4; ++it) {
      int lin = tid + 256 * it;          // 0..1023
      int c   = lin >> 5;                // 0..31 channel
      int p4  = lin & 31;                // pixel quad
      v4f f = *(const v4f*)(x + ((size_t)(b * C_ + c0 + c)) * N_ + p0 + p4 * 4);
#pragma unroll
      for (int i = 0; i < 4; ++i) Ald[(p4 * 4 + i) * 40 + c] = (_Float16)f[i];
    }
    // Stage B = weight tile (rows contiguous over input channel).
#pragma unroll
    for (int it = 0; it < 2; ++it) {
      int lin = tid + 256 * it;          // 0..511
      int o   = lin >> 3;
      int c4  = lin & 7;
      v4f f = *(const v4f*)(W + (size_t)(crel + o) * C_ + c0 + c4 * 4);
#pragma unroll
      for (int i = 0; i < 4; ++i) Bld[o * 40 + c4 * 4 + i] = (_Float16)f[i];
    }
    __syncthreads();

    v16h af = frag_from(&Ald[(wave * 16 + (lane & 15)) * 40], 0, lane);
#pragma unroll
    for (int tn = 0; tn < 4; ++tn) {
      v16h bf = frag_from(&Bld[(tn * 16 + (lane & 15)) * 40], 0, lane);
      acc[tn] = __builtin_amdgcn_wmma_f32_16x16x32_f16(false, af, false, bf,
                                                       (short)0, acc[tn], false, false);
    }
    __syncthreads();
  }

  // Bias + f16 store. C/D layout: col = lane&15, row = reg + 8*(lane>=16).
#pragma unroll
  for (int tn = 0; tn < 4; ++tn) {
    int cloc = crel + tn * 16 + (lane & 15);
    float bb = bias[cloc];
#pragma unroll
    for (int r = 0; r < 8; ++r) {
      int pix = p0 + wave * 16 + r + 8 * (lane >> 4);
      float v = acc[tn][r] + bb;
      size_t idx = vmaj ? ((size_t)(b * C_ + cloc)) * N_ + pix
                        : ((size_t)(b * N_ + pix)) * CI_ + cloc;
      dst[idx] = (_Float16)v;
    }
  }
}

// ---------------------------------------------------------------------------
// Kernel 2: flash attention + fused epilogue gamma*(softmax(QK^T)V)/l + x.
// Block: 128 query rows (1 M-tile per wave), 128 of 256 out-channels
// (blockIdx.z halves), key blocks of 64, head dim 128.
// K/V tiles are DMA'd to LDS once per block via GLOBAL_LOAD_ASYNC_TO_LDS_B128
// (instead of 8 redundant per-wave global fetches), then fragments are fed
// from LDS with ds_load_b128.
// ---------------------------------------------------------------------------
__global__ __launch_bounds__(256) void flash_attn(
    const _Float16* __restrict__ Q, const _Float16* __restrict__ K,
    const _Float16* __restrict__ V, const float* __restrict__ x,
    const float* __restrict__ gamma, float* __restrict__ out) {
  __shared__ _Float16 Klds[64 * 136];   // 64 keys x 128 ch, pitch 136 halves (17 chunks)
  __shared__ _Float16 Vlds[128 * 72];   // 128 ch x 64 keys, pitch 72 halves (9 chunks)
  __shared__ _Float16 Pld[128 * 72];    // P transpose staging, pitch 72 halves

  const int tid  = threadIdx.x;
  const int lane = tid & 31;
  const int wave = tid >> 5;
  const int m0  = blockIdx.x * 128;
  const int b   = blockIdx.y;
  const int ch0 = blockIdx.z * 128;

  const unsigned kbase = lds_off_of(&Klds[0]);
  const unsigned vbase = lds_off_of(&Vlds[0]);

  // Q fragments straight from global: (B,N,CI) rows are contiguous over c.
  const int rowq = m0 + wave * 16 + (lane & 15);
  const _Float16* qrow = Q + ((size_t)(b * N_ + rowq)) * CI_;
  v16h qf[4];
#pragma unroll
  for (int ks = 0; ks < 4; ++ks) qf[ks] = frag_from(qrow, ks * 32, lane);

  const v8f zero = {0, 0, 0, 0, 0, 0, 0, 0};
  v8f o[8];
#pragma unroll
  for (int t = 0; t < 8; ++t) o[t] = zero;
  float mrow[8], lrow[8];
#pragma unroll
  for (int r = 0; r < 8; ++r) { mrow[r] = -3.0e38f; lrow[r] = 0.0f; }

  for (int j0 = 0; j0 < N_; j0 += 64) {
    // ---- Async DMA: K tile (64 x 128 ch) and V tile (128 ch x 64 keys) ----
    // K: 64 rows x 16 b128-chunks = 1024 chunks, 4 per thread.
#pragma unroll
    for (int it = 0; it < 4; ++it) {
      int lin = tid + 256 * it;
      int row = lin >> 4, cc = lin & 15;
      async_copy_b128(kbase + (unsigned)(row * 17 + cc) * 16,
                      K + ((size_t)(b * N_ + j0 + row)) * CI_ + cc * 8);
    }
    // V: 128 rows x 8 b128-chunks = 1024 chunks, 4 per thread.
#pragma unroll
    for (int it = 0; it < 4; ++it) {
      int lin = tid + 256 * it;
      int row = lin >> 3, cc = lin & 7;
      async_copy_b128(vbase + (unsigned)(row * 9 + cc) * 16,
                      V + ((size_t)(b * C_ + ch0 + row)) * N_ + j0 + cc * 8);
    }
    asm volatile("s_wait_asynccnt 0" ::: "memory");
    __syncthreads();

    // ---- S = Q K^T for a 16 x 64 strip (4 tiles x 4 k-steps) ----
    v8f s[4];
#pragma unroll
    for (int tn = 0; tn < 4; ++tn) s[tn] = zero;
#pragma unroll
    for (int tn = 0; tn < 4; ++tn) {
      const _Float16* krow = &Klds[(tn * 16 + (lane & 15)) * 136];
#pragma unroll
      for (int ks = 0; ks < 4; ++ks) {
        v16h kf = frag_from(krow, ks * 32, lane);
        s[tn] = __builtin_amdgcn_wmma_f32_16x16x32_f16(false, qf[ks], false, kf,
                                                       (short)0, s[tn], false, false);
      }
    }

    // ---- Online softmax: rows live at (reg r, lane-half); cols across 16 lanes ----
#pragma unroll
    for (int r = 0; r < 8; ++r) {
      float v = fmaxf(fmaxf(s[0][r], s[1][r]), fmaxf(s[2][r], s[3][r]));
      v = fmaxf(v, __shfl_xor(v, 1, 32));
      v = fmaxf(v, __shfl_xor(v, 2, 32));
      v = fmaxf(v, __shfl_xor(v, 4, 32));
      v = fmaxf(v, __shfl_xor(v, 8, 32));
      float mn    = fmaxf(mrow[r], v);
      float alpha = __expf(mrow[r] - mn);
#pragma unroll
      for (int t = 0; t < 8; ++t) o[t][r] *= alpha;
      float rs = 0.f;
      int prow = (wave * 16 + r + 8 * (lane >> 4)) * 72 + (lane & 15);
#pragma unroll
      for (int tn = 0; tn < 4; ++tn) {
        float p = __expf(s[tn][r] - mn);
        Pld[prow + tn * 16] = (_Float16)p;   // C-layout -> row-major P in LDS
        rs += p;
      }
      rs += __shfl_xor(rs, 1, 32);
      rs += __shfl_xor(rs, 2, 32);
      rs += __shfl_xor(rs, 4, 32);
      rs += __shfl_xor(rs, 8, 32);
      lrow[r] = lrow[r] * alpha + rs;
      mrow[r] = mn;
    }

    // Wave-private LDS RAW: P stores must land before fragment reload.
    asm volatile("s_wait_dscnt 0" ::: "memory");

    // ---- O += P V : 2 k-steps over the 64 keys x 8 channel tiles ----
#pragma unroll
    for (int ks = 0; ks < 2; ++ks) {
      v16h pf = frag_from(&Pld[(wave * 16 + (lane & 15)) * 72], ks * 32, lane);
#pragma unroll
      for (int ct = 0; ct < 8; ++ct) {
        const _Float16* vrow = &Vlds[(ct * 16 + (lane & 15)) * 72];
        v16h vf = frag_from(vrow, ks * 32, lane);
        o[ct] = __builtin_amdgcn_wmma_f32_16x16x32_f16(false, pf, false, vf,
                                                       (short)0, o[ct], false, false);
      }
    }
    __syncthreads();   // keep next tile's DMA from overwriting live K/V reads
  }

  // ---- Epilogue: out = gamma * O / l + x (both (B,C,N) f32) ----
  const float g = gamma[0];
#pragma unroll
  for (int r = 0; r < 8; ++r) {
    float inv = 1.0f / lrow[r];
    int pix = m0 + wave * 16 + r + 8 * (lane >> 4);
#pragma unroll
    for (int ct = 0; ct < 8; ++ct) {
      int c = ch0 + ct * 16 + (lane & 15);
      size_t idx = ((size_t)(b * C_ + c)) * N_ + pix;
      out[idx] = g * (o[ct][r] * inv) + x[idx];
    }
  }
}

extern "C" void kernel_launch(void* const* d_in, const int* in_sizes, int n_in,
                              void* d_out, int out_size, void* d_ws, size_t ws_size,
                              hipStream_t stream) {
  const float* x     = (const float*)d_in[0];
  const float* Wq    = (const float*)d_in[1];
  const float* bq    = (const float*)d_in[2];
  const float* Wk    = (const float*)d_in[3];
  const float* bk    = (const float*)d_in[4];
  const float* Wv    = (const float*)d_in[5];
  const float* bv    = (const float*)d_in[6];
  const float* gamma = (const float*)d_in[7];
  float* out = (float*)d_out;

  // Workspace: Q (8MB) | K (8MB) | V (16MB), all f16.
  _Float16* Qw = (_Float16*)d_ws;
  _Float16* Kw = Qw + (size_t)B_ * N_ * CI_;
  _Float16* Vw = Kw + (size_t)B_ * N_ * CI_;

  dim3 pgrid(N_ / 128, (2 * CI_ + C_) / 64, B_);  // 32 x 8 x 8
  qkv_proj<<<pgrid, 256, 0, stream>>>(x, Wq, bq, Wk, bk, Wv, bv, Qw, Kw, Vw);

  dim3 agrid(N_ / 128, B_, 2);                    // 32 x 8 x 2 (channel halves)
  flash_attn<<<agrid, 256, 0, stream>>>(Qw, Kw, Vw, x, gamma, out);
}
